// MemorySelfAttention_8890582303066
// MI455X (gfx1250) — matrix operand order repaired and checked by reading, not compiled, not run on hardware
//
#include <hip/hip_runtime.h>
#include <hip/hip_bf16.h>

typedef __bf16 bf16_t;
typedef __attribute__((ext_vector_type(16))) __bf16 v16bf;
typedef __attribute__((ext_vector_type(8)))  float  v8f;

#define WMMA_BF16(a, b, c) \
  __builtin_amdgcn_wmma_f32_16x16x32_bf16(false, (a), false, (b), (short)0, (c), false, false)

constexpr int B_  = 2;
constexpr int T_  = 1024;
constexpr int C_  = 1024;
constexpr int H_  = 16;
constexpr int HD_ = 64;
constexpr int S_  = 512;
constexpr int L_  = 1024;
constexpr int M_  = L_ + S_ + T_;   // 2560
constexpr int ST_ = S_ + T_;        // 1536
constexpr int C3_ = 3 * C_;         // 3072

// ---- WMMA 16-bit fragment index helpers (wave32) ----
// A 16x32: lane holds row (lane&15); K packing interleaved by 8 between halves:
//   half0: V0-3 -> K0..7, V4-7 -> K16..23 ; half1: V0-3 -> K8..15, V4-7 -> K24..31
__device__ __forceinline__ int fragkA(int lane, int i) {
  int g = i >> 1;
  return ((g & 4) << 2) + ((g & 3) << 1) + (i & 1) + ((lane & 16) >> 1);
}
// B 32x16: lane holds column (lane&15); K packing interleaved by 8 between halves (same as A):
//   half0: elems 0-7 -> K0..7, 8-15 -> K16..23 ; half1: elems 0-7 -> K8..15, 8-15 -> K24..31
__device__ __forceinline__ int fragkB(int lane, int i) {
  return ((i & 8) << 1) + (i & 7) + ((lane & 16) >> 1);
}

// ---------------- Kernel 1: qkv = [stm; x] @ w_attn (bf16 WMMA, f32 out) ----
__global__ __launch_bounds__(32) void k_gemm_qkv(
    const float* __restrict__ x, const float* __restrict__ stm,
    const float* __restrict__ w_attn, float* __restrict__ qkv) {
  const int lane = threadIdx.x;
  const int tn = blockIdx.x, tm = blockIdx.y, b = blockIdx.z;
  const int lo = lane & 15, half = (lane >> 4) & 1;
  const int row = tm * 16 + lo;          // 0..1535
  const int col = tn * 16 + lo;          // 0..3071
  const float* arow = (row < S_)
      ? (stm + ((size_t)b * S_ + row) * C_)
      : (x   + ((size_t)b * T_ + (row - S_)) * C_);
  v8f acc = {};
  for (int k0 = 0; k0 < C_; k0 += 32) {
    v16bf a, w;
#pragma unroll
    for (int i = 0; i < 16; ++i) {
      a[i] = (bf16_t)arow[k0 + fragkA(lane, i)];
      w[i] = (bf16_t)w_attn[(size_t)(k0 + fragkB(lane, i)) * C3_ + col];
    }
    acc = WMMA_BF16(a, w, acc);
  }
#pragma unroll
  for (int g = 0; g < 8; ++g) {
    int m = g + 8 * half;
    qkv[((size_t)b * ST_ + tm * 16 + m) * C3_ + col] = acc[g];
  }
}

// ---------------- Kernel 2: long_k / long_v -> bf16 K/V rows [0, L) --------
__global__ void k_scatter_long(const float* __restrict__ lk,
                               const float* __restrict__ lv,
                               bf16_t* __restrict__ K, bf16_t* __restrict__ V) {
  int i = blockIdx.x * blockDim.x + threadIdx.x;
  if (i >= B_ * L_ * H_ * HD_) return;
  int d = i & 63;
  int h = (i >> 6) & 15;
  int l = (i >> 10) & 1023;
  int b = i >> 20;
  size_t dst = ((size_t)(b * H_ + h) * M_ + l) * HD_ + d;   // (B,H,M,64)
  K[dst] = (bf16_t)lk[i];                                   // (B,L,H,64) == i
  V[dst] = (bf16_t)lv[i];
}

// ---------------- Kernel 3: RoPE + scatter projected q/k/v -----------------
__global__ void k_rope_scatter(const float* __restrict__ qkv,
                               bf16_t* __restrict__ Q, bf16_t* __restrict__ K,
                               bf16_t* __restrict__ V) {
  int i = blockIdx.x * blockDim.x + threadIdx.x;
  if (i >= B_ * ST_ * H_ * (HD_ / 2)) return;
  int j = i & 31;                 // pair index 0..31
  int h = (i >> 5) & 15;
  int tmp = i >> 9;
  int r = tmp % ST_;              // rope position 0..1535
  int b = tmp / ST_;
  // inv_freq = theta^(-2j/64) = exp(-(2j/64)*ln(1e4))
  float inv = __expf(-(float)(2 * j) * (9.210340371976184f / 64.0f));
  float ang = (float)r * inv;
  float c = __cosf(ang), s = __sinf(ang);
  const float* base = qkv + ((size_t)b * ST_ + r) * C3_;
  int ch = h * HD_ + 2 * j;
  size_t kv_row = ((size_t)(b * H_ + h) * M_ + (L_ + r)) * HD_;
  float kk0 = base[C_ + ch], kk1 = base[C_ + ch + 1];
  K[kv_row + 2 * j]     = (bf16_t)(kk0 * c - kk1 * s);
  K[kv_row + 2 * j + 1] = (bf16_t)(kk0 * s + kk1 * c);
  V[kv_row + 2 * j]     = (bf16_t)base[2 * C_ + ch];
  V[kv_row + 2 * j + 1] = (bf16_t)base[2 * C_ + ch + 1];
  if (r >= S_) {                  // queries: only the last T positions matter
    float q0 = base[ch], q1 = base[ch + 1];
    size_t q_row = ((size_t)(b * H_ + h) * T_ + (r - S_)) * HD_;
    Q[q_row + 2 * j]     = (bf16_t)(q0 * c - q1 * s);
    Q[q_row + 2 * j + 1] = (bf16_t)(q0 * s + q1 * c);
  }
}

// ---------------- Kernel 4: causal flash attention (bf16 WMMA) -------------
__global__ __launch_bounds__(32) void k_attn(const bf16_t* __restrict__ Q,
                                             const bf16_t* __restrict__ K,
                                             const bf16_t* __restrict__ V,
                                             bf16_t* __restrict__ Y) {
  const int lane = threadIdx.x;
  const int lo = lane & 15, half = (lane >> 4) & 1;
  const int qt = blockIdx.x, h = blockIdx.y, b = blockIdx.z;
  __shared__ bf16_t ldsP[16 * 32];

  const bf16_t* Qb = Q + ((size_t)(b * H_ + h) * T_ + qt * 16) * HD_;
  const bf16_t* Kb = K + (size_t)(b * H_ + h) * M_ * HD_;
  const bf16_t* Vb = V + (size_t)(b * H_ + h) * M_ * HD_;

  v16bf qa0, qa1;                  // Q tile 16x64 as two A fragments
#pragma unroll
  for (int i = 0; i < 16; ++i) {
    int d = fragkA(lane, i);
    qa0[i] = Qb[(size_t)lo * HD_ + d];
    qa1[i] = Qb[(size_t)lo * HD_ + 32 + d];
  }

  v8f o0 = {}, o1 = {}, o2 = {}, o3 = {};   // O tile 16x64 (f32)
  float rmax[8], rsum[8];
#pragma unroll
  for (int g = 0; g < 8; ++g) { rmax[g] = -INFINITY; rsum[g] = 0.0f; }

  const int qglo = L_ + S_ + qt * 16;       // global index of query row 0
  const int nk = qglo + 16;                 // keys needed: 0 .. qglo+15
  const int nch = (nk + 31) / 32;

  for (int chn = 0; chn < nch; ++chn) {
    const int k0 = chn * 32;
    v8f s0 = {}, s1 = {};                   // scores: keys k0..+15, k0+16..+31
    {
      v16bf kb;
#pragma unroll
      for (int i = 0; i < 16; ++i) kb[i] = Kb[(size_t)(k0 + lo) * HD_ + fragkB(lane, i)];
      s0 = WMMA_BF16(qa0, kb, s0);
#pragma unroll
      for (int i = 0; i < 16; ++i) kb[i] = Kb[(size_t)(k0 + lo) * HD_ + 32 + fragkB(lane, i)];
      s0 = WMMA_BF16(qa1, kb, s0);
#pragma unroll
      for (int i = 0; i < 16; ++i) kb[i] = Kb[(size_t)(k0 + 16 + lo) * HD_ + fragkB(lane, i)];
      s1 = WMMA_BF16(qa0, kb, s1);
#pragma unroll
      for (int i = 0; i < 16; ++i) kb[i] = Kb[(size_t)(k0 + 16 + lo) * HD_ + 32 + fragkB(lane, i)];
      s1 = WMMA_BF16(qa1, kb, s1);
    }
    // scale + causal mask (C layout: row m = g+8*half, col n = lo)
#pragma unroll
    for (int g = 0; g < 8; ++g) {
      int qg = qglo + g + 8 * half;
      s0[g] = (k0 + lo      <= qg) ? s0[g] * 0.125f : -INFINITY;
      s1[g] = (k0 + 16 + lo <= qg) ? s1[g] * 0.125f : -INFINITY;
    }
    // online softmax (row reductions within each 16-lane half)
#pragma unroll
    for (int g = 0; g < 8; ++g) {
      float v = fmaxf(s0[g], s1[g]);
      v = fmaxf(v, __shfl_xor(v, 1, 32));
      v = fmaxf(v, __shfl_xor(v, 2, 32));
      v = fmaxf(v, __shfl_xor(v, 4, 32));
      v = fmaxf(v, __shfl_xor(v, 8, 32));
      float nm = fmaxf(rmax[g], v);
      float alpha = __expf(rmax[g] - nm);   // exp(-inf)=0 on first chunk
      rmax[g] = nm;
      float p0 = __expf(s0[g] - nm);
      float p1 = __expf(s1[g] - nm);
      float ps = p0 + p1;
      ps += __shfl_xor(ps, 1, 32);
      ps += __shfl_xor(ps, 2, 32);
      ps += __shfl_xor(ps, 4, 32);
      ps += __shfl_xor(ps, 8, 32);
      rsum[g] = rsum[g] * alpha + ps;
      o0[g] *= alpha; o1[g] *= alpha; o2[g] *= alpha; o3[g] *= alpha;
      int m = g + 8 * half;
      ldsP[m * 32 + lo]      = (bf16_t)p0;  // repack C-layout -> row-major
      ldsP[m * 32 + 16 + lo] = (bf16_t)p1;
    }
    __syncthreads();                        // single-wave WG: just orders LDS
    v16bf pa;                               // P as A fragment (16x32 over keys)
#pragma unroll
    for (int i = 0; i < 16; ++i) pa[i] = ldsP[lo * 32 + fragkA(lane, i)];
    __syncthreads();
    {
      v16bf vb;                             // V chunk 32x64 as 4 B fragments
#pragma unroll
      for (int i = 0; i < 16; ++i) vb[i] = Vb[(size_t)(k0 + fragkB(lane, i)) * HD_ + lo];
      o0 = WMMA_BF16(pa, vb, o0);
#pragma unroll
      for (int i = 0; i < 16; ++i) vb[i] = Vb[(size_t)(k0 + fragkB(lane, i)) * HD_ + 16 + lo];
      o1 = WMMA_BF16(pa, vb, o1);
#pragma unroll
      for (int i = 0; i < 16; ++i) vb[i] = Vb[(size_t)(k0 + fragkB(lane, i)) * HD_ + 32 + lo];
      o2 = WMMA_BF16(pa, vb, o2);
#pragma unroll
      for (int i = 0; i < 16; ++i) vb[i] = Vb[(size_t)(k0 + fragkB(lane, i)) * HD_ + 48 + lo];
      o3 = WMMA_BF16(pa, vb, o3);
    }
  }
  // normalize and store Y (B,T,C) bf16
#pragma unroll
  for (int g = 0; g < 8; ++g) {
    int m = g + 8 * half;
    float inv = 1.0f / rsum[g];
    size_t orow = ((size_t)b * T_ + qt * 16 + m) * C_ + h * HD_;
    Y[orow + lo]      = (bf16_t)(o0[g] * inv);
    Y[orow + 16 + lo] = (bf16_t)(o1[g] * inv);
    Y[orow + 32 + lo] = (bf16_t)(o2[g] * inv);
    Y[orow + 48 + lo] = (bf16_t)(o3[g] * inv);
  }
}

// ---------------- Kernel 5: out = Y @ w_proj (f32 out) ---------------------
__global__ __launch_bounds__(32) void k_gemm_out(const bf16_t* __restrict__ Yb,
                                                 const float* __restrict__ w_proj,
                                                 float* __restrict__ out) {
  const int lane = threadIdx.x;
  const int tn = blockIdx.x, tm = blockIdx.y;
  const int lo = lane & 15, half = (lane >> 4) & 1;
  const size_t row = (size_t)tm * 16 + lo;      // 0..2047 (B*T folded)
  const int col = tn * 16 + lo;
  v8f acc = {};
  for (int k0 = 0; k0 < C_; k0 += 32) {
    v16bf a, w;
#pragma unroll
    for (int i = 0; i < 16; ++i) {
      a[i] = Yb[row * C_ + k0 + fragkA(lane, i)];
      w[i] = (bf16_t)w_proj[(size_t)(k0 + fragkB(lane, i)) * C_ + col];
    }
    acc = WMMA_BF16(a, w, acc);
  }
#pragma unroll
  for (int g = 0; g < 8; ++g) {
    int m = g + 8 * half;
    out[((size_t)tm * 16 + m) * C_ + col] = acc[g];
  }
}

// ---------------------------------------------------------------------------
extern "C" void kernel_launch(void* const* d_in, const int* in_sizes, int n_in,
                              void* d_out, int out_size, void* d_ws, size_t ws_size,
                              hipStream_t stream) {
  (void)in_sizes; (void)n_in; (void)out_size; (void)ws_size;
  const float* x      = (const float*)d_in[0];
  const float* stm    = (const float*)d_in[1];
  // d_in[2] = long_q: never queried (only the last T query rows are output)
  const float* lk     = (const float*)d_in[3];
  const float* lv     = (const float*)d_in[4];
  const float* w_attn = (const float*)d_in[5];
  const float* w_proj = (const float*)d_in[6];
  float* out = (float*)d_out;

  char* ws = (char*)d_ws;
  float*  qkv = (float*) (ws);               // B*1536*3072 f32 = 37,748,736 B
  bf16_t* Qb  = (bf16_t*)(ws + 37748736);    // B*H*T*64 bf16   =  4,194,304 B
  bf16_t* Kb  = (bf16_t*)(ws + 41943040);    // B*H*M*64 bf16   = 10,485,760 B
  bf16_t* Vb  = (bf16_t*)(ws + 52428800);    // B*H*M*64 bf16   = 10,485,760 B
  bf16_t* Yb  = (bf16_t*)(ws + 62914560);    // B*T*C bf16      =  4,194,304 B
                                             // total: 64 MiB

  k_gemm_qkv   <<<dim3(C3_ / 16, ST_ / 16, B_), 32, 0, stream>>>(x, stm, w_attn, qkv);
  k_scatter_long<<<(B_ * L_ * H_ * HD_) / 256, 256, 0, stream>>>(lk, lv, Kb, Vb);
  k_rope_scatter<<<(B_ * ST_ * H_ * (HD_ / 2)) / 256, 256, 0, stream>>>(qkv, Qb, Kb, Vb);
  k_attn       <<<dim3(T_ / 16, H_, B_), 32, 0, stream>>>(Qb, Kb, Vb, Yb);
  k_gemm_out   <<<dim3(C_ / 16, (B_ * T_) / 16, 1), 32, 0, stream>>>(Yb, w_proj, out);
}
